// deform_conv2d_naive_84361747628508
// MI455X (gfx1250) — matrix operand order repaired and checked by reading, not compiled, not run on hardware
//
#include <hip/hip_runtime.h>
#include <hip/hip_bf16.h>

// Deformable conv2d: N=8, C=256, H=W=64, OUT_C=256, 3x3, stride 1, pad 1.
// Stage 1: NCHW->NHWC transpose of x (f32)       -> ws
// Stage 2: weight -> fp16 wTT[oc][k*256+c]       -> ws
// Stage 3: deformable im2col -> fp16 col[m][kc]  -> ws   (m = n*4096+oh*64+ow, kc = k*256+c)
// Stage 4: WMMA fp16 GEMM (128x128 block tile, double-buffered LDS):
//          out[m][oc] = sum_kc col[m][kc]*wTT[oc][kc]
// Workspace requirement: 33554432 + 1179648 + 150994944 = ~186 MB.

typedef __attribute__((ext_vector_type(16))) _Float16 v16h;
typedef __attribute__((ext_vector_type(8)))  _Float16 v8h;
typedef __attribute__((ext_vector_type(8)))  float    v8f;

#define N_IMG 8
#define CH    256
#define HH    64
#define WW    64
#define OC    256
#define KTAP  9
#define KC    2304            // 9*256
#define MTOT  32768           // 8*64*64
#define KSTEPS 72             // 2304/32
#define LDSA  5120            // 128 rows * 40 halves (pad 8)
#define LDSBUF (2 * LDSA)     // A then B per buffer

// ---------------------------------------------------------------------------
// Kernel 1: x NCHW -> NHWC (f32), padded-LDS transpose, conflict free.
// grid = N*H*(W/16) = 2048 blocks, 256 threads.
// ---------------------------------------------------------------------------
__global__ __launch_bounds__(256) void nchw2nhwc_kernel(const float* __restrict__ x,
                                                        float* __restrict__ xh) {
  __shared__ float tile[256 * 17];           // 17 KB, pad 1 float per row
  const int b      = blockIdx.x;             // ((n*64 + h)*4 + wchunk)
  const int wchunk = b & 3;
  const int h      = (b >> 2) & 63;
  const int n      = b >> 8;
  const int w0     = wchunk * 16;
  const int t      = threadIdx.x;

#pragma unroll
  for (int i = 0; i < 16; ++i) {             // load 256c x 16w, coalesced in w
    const int lin = i * 256 + t;
    const int c   = lin >> 4;
    const int w   = lin & 15;
    tile[c * 17 + w] =
        x[((((size_t)n * CH + c) * HH + h) << 6) + (size_t)(w0 + w)];
  }
  __syncthreads();
#pragma unroll
  for (int i = 0; i < 16; ++i) {             // store coalesced in c (1 KB)
    xh[((((size_t)n * HH + h) * WW) + (w0 + i)) * CH + t] = tile[t * 17 + i];
  }
}

// ---------------------------------------------------------------------------
// Kernel 2: weight[oc][c][kh][kw] (f32) -> wTT[oc][k*256+c] (f16), coalesced out.
// grid = 2304 blocks * 256 threads = 589824 = 256*2304 elements.
// ---------------------------------------------------------------------------
__global__ __launch_bounds__(256) void wxform_kernel(const float* __restrict__ w,
                                                     _Float16* __restrict__ wTT) {
  const int idx = blockIdx.x * 256 + threadIdx.x;
  const int oc  = idx / KC;
  const int kc  = idx - oc * KC;
  const int k   = kc >> 8;
  const int c   = kc & 255;
  wTT[idx] = (_Float16)w[((size_t)oc * CH + c) * KTAP + k];
}

// ---------------------------------------------------------------------------
// Kernel 3: deformable im2col. One block per output position, thread = channel.
// Bilinear coords are block-uniform => no divergence; x reads coalesced (NHWC).
// ---------------------------------------------------------------------------
__global__ __launch_bounds__(256) void im2col_kernel(const float* __restrict__ xh,
                                                     const float* __restrict__ offs,
                                                     _Float16* __restrict__ col) {
  const int b  = blockIdx.x;                 // 0..32767
  const int c  = threadIdx.x;
  const int n  = b >> 12;
  const int oh = (b >> 6) & 63;
  const int ow = b & 63;
  const float* xb = xh + ((size_t)n << 20);  // n * 64*64*256
  _Float16* colp  = col + (size_t)b * KC;

#pragma unroll
  for (int k = 0; k < KTAP; ++k) {
    const int kyi = k / 3 - 1;
    const int kxi = k % 3 - 1;
    const size_t ob = ((((size_t)n * 18 + 2 * k) * 64 + oh) * 64) + ow;
    const float offy = offs[ob];
    const float offx = offs[ob + 4096];      // channel 2k+1
    const float y  = (float)(kyi + oh) + offy;
    const float xp = (float)(kxi + ow) + offx;
    const float y0 = floorf(y);
    const float x0 = floorf(xp);
    float acc = 0.0f;
#pragma unroll
    for (int dy = 0; dy < 2; ++dy) {
#pragma unroll
      for (int dx = 0; dx < 2; ++dx) {
        const float yi = y0 + (float)dy;
        const float xi = x0 + (float)dx;
        const float wgt = (1.0f - fabsf(y - yi)) * (1.0f - fabsf(xp - xi));
        if (yi >= 0.0f && yi <= 63.0f && xi >= 0.0f && xi <= 63.0f) {
          const int yc = (int)yi;
          const int xc = (int)xi;
          acc += xb[((((size_t)yc << 6) + xc) << 8) + c] * wgt;
        }
      }
    }
    colp[k * 256 + c] = (_Float16)acc;
  }
}

// ---------------------------------------------------------------------------
// Kernel 4: WMMA GEMM, 128(M) x 128(N) block tile, 8 wave32s (4M x 2N),
// wave tile 32x64 = eight 16x16 f32 accumulators, 8 v_wmma per 32-K step.
// Double-buffered LDS (40 KB) + register staging: ONE barrier per K step;
// next tile's global_load_b128s overlap the 8 WMMAs.
// Fragment packing follows CDNA5 ISA 7.12.2 f16 A(16x32)/B(32x16) layouts.
// ---------------------------------------------------------------------------
__global__ __launch_bounds__(256) void wmma_gemm_kernel(const _Float16* __restrict__ col,
                                                        const _Float16* __restrict__ wTT,
                                                        float* __restrict__ out) {
  __shared__ alignas(16) _Float16 lds[2][LDSBUF];   // [buf][ A(5120) | B(5120) ]

  const int tid   = threadIdx.x;
  const int lane  = tid & 31;
  const int wave  = tid >> 5;
  const int mTile = blockIdx.y * 128;
  const int nTile = blockIdx.x * 128;
  const int waveM = (wave & 3) * 32;
  const int waveN = (wave >> 2) * 64;

  // Staging: thread -> two 16B chunks of A tile and two of B tile (128x32 each).
  const int ldRow  = tid >> 1;                 // 0..127
  const int ldSeg  = (tid & 1) * 16;           // 0 or 16 halves
  const size_t aBase = (size_t)(mTile + ldRow) * KC + ldSeg;
  const size_t bBase = (size_t)(nTile + ldRow) * KC + ldSeg;
  const int ldsOff = ldRow * 40 + ldSeg;

  // Fragment addressing (ISA 7.12.2):
  // A: lane row = lane&15; elems 0-7 -> K = kb..kb+7, elems 8-15 -> K = kb+16..kb+23,
  //    kb = 0 (lanes 0-15) / 8 (lanes 16-31).
  // B: lane col = lane&15; elems 0-7 -> K = ko..ko+7, elems 8-15 -> K = ko+16..ko+23, ko = 0 (lanes 0-15) / 8 (lanes 16-31).
  const int row = lane & 15;
  const int kb  = (lane < 16) ? 0 : 8;
  const int ko  = (lane < 16) ? 0 : 8;

  v8f acc[2][4];
#pragma unroll
  for (int i = 0; i < 2; ++i)
#pragma unroll
    for (int j = 0; j < 4; ++j) acc[i][j] = (v8f){};

  // Prologue: stage K-tile 0 into buffer 0.
  v8h ar0 = *(const v8h*)(col + aBase);
  v8h ar1 = *(const v8h*)(col + aBase + 8);
  v8h br0 = *(const v8h*)(wTT + bBase);
  v8h br1 = *(const v8h*)(wTT + bBase + 8);
  *(v8h*)(&lds[0][ldsOff])            = ar0;
  *(v8h*)(&lds[0][ldsOff + 8])        = ar1;
  *(v8h*)(&lds[0][LDSA + ldsOff])     = br0;
  *(v8h*)(&lds[0][LDSA + ldsOff + 8]) = br1;

  int cur = 0;
  for (int it = 0; it < KSTEPS; ++it) {
    __syncthreads();                           // buf[cur] visible; prior reads done
    const bool more = (it + 1) < KSTEPS;
    if (more) {                                // issue next tile's global loads now
      const size_t k = (size_t)(it + 1) * 32;
      ar0 = *(const v8h*)(col + aBase + k);
      ar1 = *(const v8h*)(col + aBase + k + 8);
      br0 = *(const v8h*)(wTT + bBase + k);
      br1 = *(const v8h*)(wTT + bBase + k + 8);
    }

    const _Float16* As = &lds[cur][0];
    const _Float16* Bs = &lds[cur][LDSA];

    v16h a[2], b[4];
#pragma unroll
    for (int i = 0; i < 2; ++i) {
      const _Float16* ap = As + (waveM + 16 * i + row) * 40 + kb;
      const v8h lo = *(const v8h*)(ap);
      const v8h hi = *(const v8h*)(ap + 16);
#pragma unroll
      for (int e = 0; e < 8; ++e) { a[i][e] = lo[e]; a[i][e + 8] = hi[e]; }
    }
#pragma unroll
    for (int j = 0; j < 4; ++j) {
      const _Float16* bp = Bs + (waveN + 16 * j + row) * 40 + ko;
      const v8h lo = *(const v8h*)(bp);
      const v8h hi = *(const v8h*)(bp + 16);
#pragma unroll
      for (int e = 0; e < 8; ++e) { b[j][e] = lo[e]; b[j][e + 8] = hi[e]; }
    }

#pragma unroll
    for (int i = 0; i < 2; ++i)
#pragma unroll
      for (int j = 0; j < 4; ++j)
        acc[i][j] = __builtin_amdgcn_wmma_f32_16x16x32_f16(
            false, a[i], false, b[j], (short)0, acc[i][j], false, false);

    if (more) {                                // stage next tile into other buffer
      _Float16* An = &lds[cur ^ 1][0];
      _Float16* Bn = &lds[cur ^ 1][LDSA];
      *(v8h*)(An + ldsOff)     = ar0;
      *(v8h*)(An + ldsOff + 8) = ar1;
      *(v8h*)(Bn + ldsOff)     = br0;
      *(v8h*)(Bn + ldsOff + 8) = br1;
    }
    cur ^= 1;
  }

  // C/D layout: VGPR r, lanes 0-15 -> M=r, N=lane; lanes 16-31 -> M=r+8, N=lane-16.
  const int mRow = mTile + waveM + ((lane >> 4) << 3);
  const int nc   = nTile + waveN + (lane & 15);
#pragma unroll
  for (int i = 0; i < 2; ++i) {
#pragma unroll
    for (int j = 0; j < 4; ++j) {
#pragma unroll
      for (int r = 0; r < 8; ++r) {
        const int M = mRow + 16 * i + r;
        const size_t pos  = (size_t)(M & 4095);
        const size_t nimg = (size_t)(M >> 12);
        out[((nimg * OC + (size_t)(nc + 16 * j)) << 12) + pos] = acc[i][j][r];
      }
    }
  }
}

// ---------------------------------------------------------------------------
extern "C" void kernel_launch(void* const* d_in, const int* in_sizes, int n_in,
                              void* d_out, int out_size, void* d_ws, size_t ws_size,
                              hipStream_t stream) {
  const float* x    = (const float*)d_in[0];   // (8,256,64,64) f32
  const float* offs = (const float*)d_in[1];   // (8,18,64,64)  f32
  const float* wgt  = (const float*)d_in[2];   // (256,256,3,3) f32
  float* out        = (float*)d_out;           // (8,256,64,64) f32

  char* ws = (char*)d_ws;
  float*    xh  = (float*)ws;                                    // 33,554,432 B
  _Float16* wTT = (_Float16*)(ws + 33554432);                    //  1,179,648 B
  _Float16* col = (_Float16*)(ws + 33554432 + 1179648);          // 150,994,944 B

  nchw2nhwc_kernel<<<2048, 256, 0, stream>>>(x, xh);
  wxform_kernel<<<2304, 256, 0, stream>>>(wgt, wTT);
  im2col_kernel<<<32768, 256, 0, stream>>>(xh, offs, col);
  wmma_gemm_kernel<<<dim3(OC / 128, MTOT / 128), 256, 0, stream>>>(col, wTT, out);
}
